// TransformerBlock_154618823216
// MI455X (gfx1250) — compile-verified, matrix operand order checked by reading, not run on hardware
//
#include <hip/hip_runtime.h>
#include <cstdint>

// ---------------------------------------------------------------------------
// Types / helpers
// ---------------------------------------------------------------------------
typedef unsigned short u16;
typedef __attribute__((ext_vector_type(16))) __bf16 v16bf;
typedef __attribute__((ext_vector_type(8)))  float  v8f;

#if __has_builtin(__builtin_amdgcn_global_load_async_to_lds_b128)
#define HAVE_ASYNC_LDS 1
#else
#define HAVE_ASYNC_LDS 0
#endif

// The async-to-LDS builtin signature (from compiler diagnostics):
//   (addrspace(1) v4i*, addrspace(3) v4i*, imm int offset, imm int cpol)
typedef int v4i_gcc __attribute__((vector_size(16)));

// Flat shared-pointer low 32 bits == LDS byte offset (aperture rule), so an
// integer round-trip produces a valid addrspace(3) pointer.
static __device__ __forceinline__ __attribute__((address_space(3))) v4i_gcc*
as_lds128(void* p) {
  return (__attribute__((address_space(3))) v4i_gcc*)(unsigned)(uintptr_t)p;
}
static __device__ __forceinline__ __attribute__((address_space(1))) v4i_gcc*
as_gbl128(const void* p) {
  return (__attribute__((address_space(1))) v4i_gcc*)(uintptr_t)p;
}

static __device__ __forceinline__ void wait_async0() {
#if HAVE_ASYNC_LDS
#if __has_builtin(__builtin_amdgcn_s_wait_asynccnt)
  __builtin_amdgcn_s_wait_asynccnt(0);
#else
  asm volatile("s_wait_asynccnt 0" ::: "memory");
#endif
#endif
}

static __device__ __forceinline__ u16 f2bf(float f) {
  unsigned u = __float_as_uint(f);
  u += 0x7fffu + ((u >> 16) & 1u);      // round-to-nearest-even
  return (u16)(u >> 16);
}

// Load one 16-bit WMMA fragment half-pair: elements [0..7] from p[0..7],
// elements [8..15] from p[16..23]  (matches CDNA5 16-bit A/B VGPR layout when
// the caller picks row = lane%16 and kbase = (lane/16)*8).
static __device__ __forceinline__ v16bf load_frag(const u16* p) {
  uint4 a = *(const uint4*)(p);
  uint4 b = *(const uint4*)(p + 16);
  v16bf v;
  __builtin_memcpy(&v, &a, 16);
  __builtin_memcpy((char*)&v + 16, &b, 16);
  return v;
}

static __device__ __forceinline__ v8f wmma_bf16(v16bf a, v16bf b, v8f c) {
  return __builtin_amdgcn_wmma_f32_16x16x32_bf16(false, a, false, b,
                                                 (short)0, c, false, false);
}

static __device__ __forceinline__ float gelu_f(float x) {
  return 0.5f * x * (1.0f + tanhf(0.7978845608028654f * (x + 0.044715f * x * x * x)));
}

// ---------------------------------------------------------------------------
// Weight convert + transpose: WT[n*K + k] = bf16(W[k*N + n])
// ---------------------------------------------------------------------------
__global__ __launch_bounds__(256) void convT_kernel(const float* __restrict__ W,
                                                    u16* __restrict__ WT,
                                                    int K, int N) {
  size_t total = (size_t)K * N;
  for (size_t i = (size_t)blockIdx.x * blockDim.x + threadIdx.x; i < total;
       i += (size_t)gridDim.x * blockDim.x) {
    size_t k = i / (size_t)N;
    size_t n = i % (size_t)N;
    WT[n * (size_t)K + k] = f2bf(W[k * (size_t)N + n]);
  }
}

// ---------------------------------------------------------------------------
// LayerNorm (unbiased variance, matches torch .var(): /(D-1)), fp32 -> bf16
// One 256-thread block per row; D == 1024 (4 elements / thread).
// ---------------------------------------------------------------------------
__global__ __launch_bounds__(256) void ln_kernel(const float* __restrict__ x,
                                                 const float* __restrict__ sc,
                                                 const float* __restrict__ sh,
                                                 u16* __restrict__ out, int D) {
  const int row = blockIdx.x;
  const int tid = threadIdx.x;
  const float* xr = x + (size_t)row * D;

  float v[4];
  float s = 0.f, ss = 0.f;
#pragma unroll
  for (int j = 0; j < 4; ++j) {
    v[j] = xr[tid + 256 * j];
    s += v[j];
    ss += v[j] * v[j];
  }

  __shared__ float r1[256];
  __shared__ float r2[256];
  r1[tid] = s; r2[tid] = ss;
  __syncthreads();
  for (int off = 128; off > 0; off >>= 1) {
    if (tid < off) { r1[tid] += r1[tid + off]; r2[tid] += r2[tid + off]; }
    __syncthreads();
  }
  const float mean = r1[0] / (float)D;
  const float var  = (r2[0] - (float)D * mean * mean) / (float)(D - 1);
  const float rs   = rsqrtf(var + 1e-6f);

#pragma unroll
  for (int j = 0; j < 4; ++j) {
    int i = tid + 256 * j;
    out[(size_t)row * D + i] = f2bf(sc[i] * (v[j] - mean) * rs + sh[i]);
  }
}

// ---------------------------------------------------------------------------
// bf16 WMMA GEMM:  out[M,N] = A[M,K] * Bt[N,K]^T  (+bias)(+gelu)(+res)
// Block tile 128x128, K-step 64. Double-buffered LDS fed by
// GLOBAL_LOAD_ASYNC_TO_LDS_B128 (ASYNCcnt) when available, overlapping the
// next tile's DMA with the current tile's 16 WMMAs. 256 threads = 8 waves,
// wave tile 32(M) x 64(N) = 2x4 WMMA tiles.
// ---------------------------------------------------------------------------
template <bool HAS_BIAS, bool DO_GELU, bool HAS_RES, bool OUT_BF16>
__global__ __launch_bounds__(256) void gemm_bf16_kernel(
    const u16* __restrict__ A, const u16* __restrict__ Bt,
    const float* __restrict__ bias, const float* __restrict__ res,
    void* __restrict__ outp, int M, int N, int K) {
  __shared__ u16 As[2][128][72];   // 72 = 64 + 8 pad (144B rows keep 16B align)
  __shared__ u16 Bs[2][128][72];

  const int tid  = threadIdx.x;
  const int lane = tid & 31;
  const int wid  = tid >> 5;
  const int wm   = wid & 3;            // 0..3 : 32-row group
  const int wn   = wid >> 2;           // 0..1 : 64-col group
  const int lr   = lane & 15;
  const int hb8  = (lane >> 4) * 8;

  const int m0 = blockIdx.y * 128;
  const int n0 = blockIdx.x * 128;

  const v8f vzero = {0.f, 0.f, 0.f, 0.f, 0.f, 0.f, 0.f, 0.f};
  v8f acc[2][4];
#pragma unroll
  for (int i = 0; i < 2; ++i)
#pragma unroll
    for (int j = 0; j < 4; ++j) acc[i][j] = vzero;

  // issue one 128x64 bf16 tile pair into LDS buffer `bufi`
  auto issue = [&](int bufi, int k0) {
#pragma unroll
    for (int i = 0; i < 4; ++i) {
      int c  = tid + 256 * i;
      int r  = c >> 3;
      int c8 = (c & 7) * 8;
#if HAVE_ASYNC_LDS
      __builtin_amdgcn_global_load_async_to_lds_b128(
          as_gbl128(&A[(size_t)(m0 + r) * K + k0 + c8]),
          as_lds128(&As[bufi][r][c8]), 0, 0);
      __builtin_amdgcn_global_load_async_to_lds_b128(
          as_gbl128(&Bt[(size_t)(n0 + r) * K + k0 + c8]),
          as_lds128(&Bs[bufi][r][c8]), 0, 0);
#else
      *(uint4*)&As[bufi][r][c8] = *(const uint4*)&A[(size_t)(m0 + r) * K + k0 + c8];
      *(uint4*)&Bs[bufi][r][c8] = *(const uint4*)&Bt[(size_t)(n0 + r) * K + k0 + c8];
#endif
    }
  };

  issue(0, 0);
  int buf = 0;

  for (int k0 = 0; k0 < K; k0 += 64) {
    wait_async0();        // tile `buf` landed in LDS (async path)
    __syncthreads();      // ...and every wave is past its reads of buf^1
    if (k0 + 64 < K) issue(buf ^ 1, k0 + 64);

#pragma unroll
    for (int ks = 0; ks < 2; ++ks) {
      const int kb = ks * 32 + hb8;
      v16bf af[2], bfr[4];
#pragma unroll
      for (int tm = 0; tm < 2; ++tm)
        af[tm] = load_frag(&As[buf][wm * 32 + tm * 16 + lr][kb]);
#pragma unroll
      for (int tn = 0; tn < 4; ++tn)
        bfr[tn] = load_frag(&Bs[buf][wn * 64 + tn * 16 + lr][kb]);
#pragma unroll
      for (int tm = 0; tm < 2; ++tm)
#pragma unroll
        for (int tn = 0; tn < 4; ++tn)
          acc[tm][tn] = wmma_bf16(af[tm], bfr[tn], acc[tm][tn]);
    }
    buf ^= 1;
  }

  // Epilogue. f32 C/D layout: VGPR g -> row (g + 8*(lane>=16)), col = lane%16.
#pragma unroll
  for (int tm = 0; tm < 2; ++tm) {
#pragma unroll
    for (int tn = 0; tn < 4; ++tn) {
      const int col = n0 + wn * 64 + tn * 16 + lr;
      const float bsv = HAS_BIAS ? bias[col] : 0.0f;
#pragma unroll
      for (int g = 0; g < 8; ++g) {
        const int row = m0 + wm * 32 + tm * 16 + hb8 + g;
        float v = acc[tm][tn][g];
        if (HAS_BIAS) v += bsv;
        if (DO_GELU) v = gelu_f(v);
        if (HAS_RES) v += res[(size_t)row * N + col];
        if (OUT_BF16) ((u16*)outp)[(size_t)row * N + col] = f2bf(v);
        else          ((float*)outp)[(size_t)row * N + col] = v;
      }
    }
  }
}

// ---------------------------------------------------------------------------
// Causal flash-attention, one wave per 16 query rows per (b,h). HD == 64.
// q/k/v/ctx are bf16 [B*S, D] with head h at columns [h*64, h*64+64).
// ---------------------------------------------------------------------------
__global__ __launch_bounds__(32) void attn_kernel(const u16* __restrict__ qb,
                                                  const u16* __restrict__ kb,
                                                  const u16* __restrict__ vb,
                                                  u16* __restrict__ ctx,
                                                  int S, int D, int HD) {
  __shared__ u16 Pl[16][32];   // probs 16x32, re-laid-out for A-fragment
  __shared__ u16 Vt[64][32];   // V tile transposed: [d][key_local]

  const int lane = threadIdx.x;
  const int lr   = lane & 15;
  const int hb8  = (lane >> 4) * 8;
  const int q0   = blockIdx.x * 16;
  const int h    = blockIdx.y;
  const int b    = blockIdx.z;

  const size_t base = (size_t)b * S * D + (size_t)h * HD;

  // Q fragments (16x64 split into two 16x32 A-fragments)
  const v16bf qf0 = load_frag(&qb[base + (size_t)(q0 + lr) * D + 0  + hb8]);
  const v16bf qf1 = load_frag(&qb[base + (size_t)(q0 + lr) * D + 32 + hb8]);

  const v8f vzero = {0.f, 0.f, 0.f, 0.f, 0.f, 0.f, 0.f, 0.f};
  v8f o[4];
#pragma unroll
  for (int t = 0; t < 4; ++t) o[t] = vzero;
  float mrun[8], lsum[8];
#pragma unroll
  for (int g = 0; g < 8; ++g) { mrun[g] = -1e30f; lsum[g] = 0.f; }

  const int nk = (q0 + 16 + 31) / 32;   // causal: only keys <= q0+15 needed
  for (int it = 0; it < nk; ++it) {
    const int kt = it * 32;

    // V tile 32(keys) x 64(d) -> LDS transposed (lane owns one key row)
    {
      const u16* vrow = &vb[base + (size_t)(kt + lane) * D];
#pragma unroll
      for (int d8 = 0; d8 < 64; d8 += 8) {
        uint4 t4 = *(const uint4*)(vrow + d8);
        u16 tmp[8];
        __builtin_memcpy(tmp, &t4, 16);
#pragma unroll
        for (int j = 0; j < 8; ++j) Vt[d8 + j][lane] = tmp[j];
      }
    }

    // scores: two 16x16 tiles over keys [kt, kt+32)
    v8f s[2];
#pragma unroll
    for (int t = 0; t < 2; ++t) {
      const int kt2 = kt + t * 16;
      v16bf kf0 = load_frag(&kb[base + (size_t)(kt2 + lr) * D + 0  + hb8]);
      v16bf kf1 = load_frag(&kb[base + (size_t)(kt2 + lr) * D + 32 + hb8]);
      v8f z = vzero;
      z = wmma_bf16(qf0, kf0, z);
      z = wmma_bf16(qf1, kf1, z);
      s[t] = z;
    }

    // scale + causal mask + online softmax (row = q0 + hb8 + g, col = lane%16)
#pragma unroll
    for (int g = 0; g < 8; ++g) {
      const int row = q0 + hb8 + g;
#pragma unroll
      for (int t = 0; t < 2; ++t) {
        const int col = kt + t * 16 + lr;
        float v = s[t][g] * 0.125f;           // 1/sqrt(64)
        if (col > row) v = -1e30f;
        s[t][g] = v;
      }
      float mx = fmaxf(s[0][g], s[1][g]);
      mx = fmaxf(mx, __shfl_xor(mx, 1));
      mx = fmaxf(mx, __shfl_xor(mx, 2));
      mx = fmaxf(mx, __shfl_xor(mx, 4));
      mx = fmaxf(mx, __shfl_xor(mx, 8));
      const float mn = fmaxf(mrun[g], mx);
      const float al = __expf(mrun[g] - mn);
      mrun[g] = mn;
      const float p0 = __expf(s[0][g] - mn);
      const float p1 = __expf(s[1][g] - mn);
      float rs = p0 + p1;
      rs += __shfl_xor(rs, 1);
      rs += __shfl_xor(rs, 2);
      rs += __shfl_xor(rs, 4);
      rs += __shfl_xor(rs, 8);
      lsum[g] = lsum[g] * al + rs;
#pragma unroll
      for (int t = 0; t < 4; ++t) o[t][g] *= al;
      Pl[hb8 + g][lr]      = f2bf(p0);
      Pl[hb8 + g][16 + lr] = f2bf(p1);
    }
    __syncthreads();

    // P(16x32) as A-fragment, V^T columns as B-fragments: O += P * V
    const v16bf pf = load_frag(&Pl[lr][hb8]);
#pragma unroll
    for (int t = 0; t < 4; ++t) {
      v16bf vf = load_frag(&Vt[t * 16 + lr][hb8]);
      o[t] = wmma_bf16(pf, vf, o[t]);
    }
    __syncthreads();
  }

  // finalize: ctx = O / l
#pragma unroll
  for (int t = 0; t < 4; ++t) {
#pragma unroll
    for (int g = 0; g < 8; ++g) {
      const int row = q0 + hb8 + g;
      const int col = t * 16 + lr;
      ctx[base + (size_t)row * D + col] = f2bf(o[t][g] / lsum[g]);
    }
  }
}

// ---------------------------------------------------------------------------
// Host launch
// ---------------------------------------------------------------------------
extern "C" void kernel_launch(void* const* d_in, const int* in_sizes, int n_in,
                              void* d_out, int out_size, void* d_ws, size_t ws_size,
                              hipStream_t stream) {
  (void)in_sizes; (void)n_in; (void)out_size; (void)ws_size;
  constexpr int Bb = 2, S = 2048, D = 1024, Hh = 16, HD = 64;
  constexpr int M  = Bb * S;    // 4096
  constexpr int D4 = 4 * D;     // 4096

  const float* x   = (const float*)d_in[0];
  const float* Wq  = (const float*)d_in[1];
  const float* Wk  = (const float*)d_in[2];
  const float* Wv  = (const float*)d_in[3];
  const float* Wo  = (const float*)d_in[4];
  const float* bo  = (const float*)d_in[5];
  const float* W1  = (const float*)d_in[6];
  const float* b1  = (const float*)d_in[7];
  const float* W2  = (const float*)d_in[8];
  const float* b2  = (const float*)d_in[9];
  const float* sc1 = (const float*)d_in[10];
  const float* sh1 = (const float*)d_in[11];
  const float* sc2 = (const float*)d_in[12];
  const float* sh2 = (const float*)d_in[13];

  char* p = (char*)d_ws;
  auto alloc = [&](size_t bytes) -> void* {
    void* r = (void*)p;
    p += (bytes + 255) & ~(size_t)255;
    return r;
  };

  u16* WqT  = (u16*)alloc((size_t)D * D * 2);
  u16* WkT  = (u16*)alloc((size_t)D * D * 2);
  u16* WvT  = (u16*)alloc((size_t)D * D * 2);
  u16* WoT  = (u16*)alloc((size_t)D * D * 2);
  u16* W1T  = (u16*)alloc((size_t)D4 * D * 2);   // [N=4096][K=1024]
  u16* W2T  = (u16*)alloc((size_t)D * D4 * 2);   // [N=1024][K=4096]
  u16* ln1  = (u16*)alloc((size_t)M * D * 2);
  u16* qb   = (u16*)alloc((size_t)M * D * 2);
  u16* kb   = (u16*)alloc((size_t)M * D * 2);
  u16* vb   = (u16*)alloc((size_t)M * D * 2);
  u16* ctxb = (u16*)alloc((size_t)M * D * 2);
  u16* ln2  = (u16*)alloc((size_t)M * D * 2);
  u16* ffn1 = (u16*)alloc((size_t)M * D4 * 2);
  float* hb = (float*)alloc((size_t)M * D * 4);

  // 1) weight convert + transpose (fp32 -> bf16, N-major)
  convT_kernel<<<1024, 256, 0, stream>>>(Wq, WqT, D, D);
  convT_kernel<<<1024, 256, 0, stream>>>(Wk, WkT, D, D);
  convT_kernel<<<1024, 256, 0, stream>>>(Wv, WvT, D, D);
  convT_kernel<<<1024, 256, 0, stream>>>(Wo, WoT, D, D);
  convT_kernel<<<2048, 256, 0, stream>>>(W1, W1T, D, D4);
  convT_kernel<<<2048, 256, 0, stream>>>(W2, W2T, D4, D);

  // 2) ln1 = LN(x)
  ln_kernel<<<M, 256, 0, stream>>>(x, sc1, sh1, ln1, D);

  // 3) q,k,v = ln1 @ W{q,k,v}
  dim3 g1(D / 128, M / 128);
  gemm_bf16_kernel<false, false, false, true><<<g1, 256, 0, stream>>>(ln1, WqT, nullptr, nullptr, qb, M, D, D);
  gemm_bf16_kernel<false, false, false, true><<<g1, 256, 0, stream>>>(ln1, WkT, nullptr, nullptr, kb, M, D, D);
  gemm_bf16_kernel<false, false, false, true><<<g1, 256, 0, stream>>>(ln1, WvT, nullptr, nullptr, vb, M, D, D);

  // 4) causal flash attention
  attn_kernel<<<dim3(S / 16, Hh, Bb), 32, 0, stream>>>(qb, kb, vb, ctxb, S, D, HD);

  // 5) h = x + ctx @ Wo + bo   (fp32)
  gemm_bf16_kernel<true, false, true, false><<<g1, 256, 0, stream>>>(ctxb, WoT, bo, x, hb, M, D, D);

  // 6) ln2 = LN(h)
  ln_kernel<<<M, 256, 0, stream>>>(hb, sc2, sh2, ln2, D);

  // 7) ffn1 = gelu(ln2 @ W1 + b1)   (bf16)
  dim3 g2(D4 / 128, M / 128);
  gemm_bf16_kernel<true, true, false, true><<<g2, 256, 0, stream>>>(ln2, W1T, b1, nullptr, ffn1, M, D4, D);

  // 8) out = h + ffn1 @ W2 + b2   (fp32, directly into d_out)
  gemm_bf16_kernel<true, false, true, false><<<g1, 256, 0, stream>>>(ffn1, W2T, b2, hb, (float*)d_out, M, D, D4);
}
